// EvalMultiModel_47768626266563
// MI455X (gfx1250) — matrix operand order repaired and checked by reading, not compiled, not run on hardware
//
#include <hip/hip_runtime.h>
#include <hip/hip_bf16.h>
#include <math.h>

// ---------------- problem constants (match reference) ----------------
#define MM      2
#define DD      128
#define RR      8
#define N0C     8192
#define LL      32
#define KK      4096
#define N_THAX  1000
#define NTOT    (N0C + LL * KK)      // 139264
#define TMAX    264                  // max 16-node tiles per level (K/16 + R-1 pad tiles)
#define KPAD    (TMAX * 16)          // 4224

typedef __attribute__((ext_vector_type(16))) __bf16 v16bf;
typedef __attribute__((ext_vector_type(8)))  float  v8f;

union ABfrag {
    unsigned int u[8];
    uint4        q[2];
    v16bf        v;
};

__device__ __forceinline__ unsigned short f2bf(float f) {
    unsigned int u = __float_as_uint(f);
    unsigned int r = (u + 0x7FFFu + ((u >> 16) & 1u)) >> 16;   // RNE
    return (unsigned short)r;
}
__device__ __forceinline__ float bf2f(unsigned short h) {
    return __uint_as_float(((unsigned int)h) << 16);
}

// ---------------- weight conversion: f32 [mr][2D][D] -> bf16 [mr][D][2D] (transposed, N-major)
__global__ __launch_bounds__(256) void cvt_w1_kernel(const float* __restrict__ W1,
                                                     unsigned short* __restrict__ W1t) {
    int i = blockIdx.x * 256 + threadIdx.x;          // over M*R*2D*D = 524288
    if (i >= MM * RR * 2 * DD * DD) return;
    int k   = i % (2 * DD);
    int col = (i / (2 * DD)) % DD;
    int mr  = i / (2 * DD * DD);
    W1t[i] = f2bf(W1[((size_t)mr * 2 * DD + k) * DD + col]);
}
__global__ __launch_bounds__(256) void cvt_w2_kernel(const float* __restrict__ W2,
                                                     unsigned short* __restrict__ W2t) {
    int i = blockIdx.x * 256 + threadIdx.x;          // over M*R*D*D = 262144
    if (i >= MM * RR * DD * DD) return;
    int k   = i % DD;
    int col = (i / DD) % DD;
    int mr  = i / (DD * DD);
    W2t[i] = f2bf(W2[((size_t)mr * DD + k) * DD + col]);
}

// ---------------- init nodes: embedding lookup * sigmoid(sine) -> bf16 store
__global__ __launch_bounds__(256) void init_kernel(const int* __restrict__ thax_ids,
                                                   const int* __restrict__ sine_ids,
                                                   const float* __restrict__ thax_table,
                                                   const float* __restrict__ sine_w,
                                                   const float* __restrict__ sine_b,
                                                   unsigned short* __restrict__ store) {
    int i = blockIdx.x * 256 + threadIdx.x;          // over N0*D
    if (i >= N0C * DD) return;
    int n = i >> 7, d = i & (DD - 1);
    int th = thax_ids[n];
    float s = (float)sine_ids[n];
#pragma unroll
    for (int m = 0; m < MM; ++m) {
        float e = thax_table[((size_t)m * N_THAX + th) * DD + d];
        float z = s * sine_w[m * DD + d] + sine_b[m * DD + d];
        float g = 1.0f / (1.0f + expf(-z));
        store[((size_t)n * MM + m) * DD + d] = f2bf(e * g);
    }
}

// ---------------- per-level rule bucketing (segments padded to 16, sentinel -1)
__global__ __launch_bounds__(256) void bucket_kernel(const int* __restrict__ rule_ids,
                                                     int* __restrict__ sorted,
                                                     int* __restrict__ tile_rule) {
    int l = blockIdx.x;
    int t = threadIdx.x;
    __shared__ int cnt[RR], cur[RR];
    if (t < RR) cnt[t] = 0;
    for (int i = t; i < KPAD; i += 256) sorted[l * KPAD + i] = -1;
    for (int i = t; i < TMAX; i += 256) tile_rule[l * TMAX + i] = -1;
    __syncthreads();
    for (int k = t; k < KK; k += 256) atomicAdd(&cnt[rule_ids[l * KK + k]], 1);
    __syncthreads();
    if (t == 0) {
        int start = 0;
        for (int r = 0; r < RR; ++r) {
            cur[r] = start;
            int tiles = (cnt[r] + 15) >> 4;
            int tb = start >> 4;
            for (int q = 0; q < tiles; ++q) tile_rule[l * TMAX + tb + q] = r;
            start += tiles << 4;
        }
    }
    __syncthreads();
    for (int k = t; k < KK; k += 256) {
        int r = rule_ids[l * KK + k];
        int pos = atomicAdd(&cur[r], 1);
        sorted[l * KPAD + pos] = k;
    }
}

// ---------------- level kernel: 16-node x 1-member tile, 4 waves, bf16 WMMA MLP
__global__ __launch_bounds__(128) void level_kernel(unsigned short* __restrict__ store,
                                                    const unsigned short* __restrict__ W1t,
                                                    const unsigned short* __restrict__ W2t,
                                                    const float* __restrict__ b1,
                                                    const float* __restrict__ b2,
                                                    const int* __restrict__ sorted,
                                                    const int* __restrict__ tile_rule,
                                                    const int* __restrict__ parents,
                                                    int l) {
    const int tile = blockIdx.x;
    const int m    = blockIdx.y;
    const int r    = tile_rule[l * TMAX + tile];
    if (r < 0) return;                                // uniform per block

    __shared__ int nds[16];
    __shared__ unsigned int x32[16 * 128];            // x tile [16][256] bf16 (8 KB)
    __shared__ unsigned short h16[16 * 128];          // h tile [16][128] bf16 (4 KB)

    const int t = threadIdx.x;
    if (t < 16) nds[t] = sorted[l * KPAD + tile * 16 + t];
    __syncthreads();

    // cooperative gather of both parents' embeddings into x tile (row-major, K contiguous)
    {
        int row = t >> 3, seg = t & 7;
        int loc = nds[row]; if (loc < 0) loc = 0;     // pad rows compute on node0's parents
        int p = parents[((size_t)l * KK + loc) * 2 + (seg >> 2)];
        const uint4* src = (const uint4*)(store + ((size_t)p * MM + m) * DD + (seg & 3) * 32);
        uint4* dst = (uint4*)&x32[row * 128 + seg * 16];
        dst[0] = src[0]; dst[1] = src[1]; dst[2] = src[2]; dst[3] = src[3];
    }
    __syncthreads();

    const int lane = t & 31;
    const int wv   = t >> 5;
    const int lrow = lane & 15;
    const int hi   = lane >> 4;        // lane half selects M/K sub-block per ISA layout
    const int adw  = hi * 4;           // A-fragment dword base (K base = hi*8)
    const size_t mrD = (size_t)(m * RR + r) * DD;

    // ---- GEMM1: h = relu(x[16,256] * W1[256,128] + b1), each wave owns 32 cols
    for (int q = 0; q < 2; ++q) {
        int nt  = wv * 2 + q;
        int col = nt * 16 + lrow;
        v8f acc = {};
        const unsigned short* wrow = W1t + (mrD + col) * (2 * DD);
#pragma unroll
        for (int ks = 0; ks < 8; ++ks) {
            ABfrag a, b;
#pragma unroll
            for (int j = 0; j < 4; ++j) {
                a.u[j]     = x32[lrow * 128 + ks * 16 + adw + j];        // K = base+0..7
                a.u[4 + j] = x32[lrow * 128 + ks * 16 + 8 + adw + j];    // K = base+16..23
            }
            const uint4* wp = (const uint4*)(wrow + ks * 32 + hi * 8);   // K = base+0..7, base+16..23
            b.q[0] = wp[0]; b.q[1] = wp[2];
            acc = __builtin_amdgcn_wmma_f32_16x16x32_bf16(
                false, a.v, false, b.v, (short)0, acc, false, false);
        }
        float bias = b1[mrD + col];
#pragma unroll
        for (int i = 0; i < 8; ++i) {
            float hv = acc[i] + bias;
            hv = hv > 0.0f ? hv : 0.0f;
            h16[(i + hi * 8) * 128 + col] = f2bf(hv);
        }
    }
    __syncthreads();

    // ---- GEMM2: out = h[16,128] * W2[128,128] + b2, scatter to store
    const unsigned int* h32 = (const unsigned int*)h16;
    const size_t gbase = ((size_t)(N0C + l * KK)) * MM;
    for (int q = 0; q < 2; ++q) {
        int nt  = wv * 2 + q;
        int col = nt * 16 + lrow;
        v8f acc = {};
        const unsigned short* wrow = W2t + (mrD + col) * DD;
#pragma unroll
        for (int ks = 0; ks < 4; ++ks) {
            ABfrag a, b;
#pragma unroll
            for (int j = 0; j < 4; ++j) {
                a.u[j]     = h32[lrow * 64 + ks * 16 + adw + j];
                a.u[4 + j] = h32[lrow * 64 + ks * 16 + 8 + adw + j];
            }
            const uint4* wp = (const uint4*)(wrow + ks * 32 + hi * 8);
            b.q[0] = wp[0]; b.q[1] = wp[2];
            acc = __builtin_amdgcn_wmma_f32_16x16x32_bf16(
                false, a.v, false, b.v, (short)0, acc, false, false);
        }
        float bias = b2[mrD + col];
#pragma unroll
        for (int i = 0; i < 8; ++i) {
            int loc = nds[i + hi * 8];
            if (loc >= 0) {                             // mask only the final stores
                store[(gbase + (size_t)loc * MM + m) * DD + col] = f2bf(acc[i] + bias);
            }
        }
    }
}

// ---------------- global pos/neg count sums (for pos_weight)
__global__ __launch_bounds__(256) void sums_kernel(const float* __restrict__ pos_cnt,
                                                   const float* __restrict__ neg_cnt,
                                                   float* __restrict__ scal) {
    __shared__ float rp, rn;
    int t = threadIdx.x;
    if (t == 0) { rp = 0.0f; rn = 0.0f; }
    __syncthreads();
    int n = blockIdx.x * 256 + t;
    float p = (n < NTOT) ? pos_cnt[n] : 0.0f;
    float q = (n < NTOT) ? neg_cnt[n] : 0.0f;
    if (p != 0.0f) atomicAdd(&rp, p);
    if (q != 0.0f) atomicAdd(&rn, q);
    __syncthreads();
    if (t == 0) { atomicAdd(&scal[0], rp); atomicAdd(&scal[1], rn); }
}

// ---------------- fused eval + weighted-BCE loss reduction
__global__ __launch_bounds__(256) void eval_kernel(const unsigned short* __restrict__ store,
                                                   const float* __restrict__ w_eval,
                                                   const float* __restrict__ b_eval,
                                                   const float* __restrict__ pos_cnt,
                                                   const float* __restrict__ neg_cnt,
                                                   const float* __restrict__ scal,
                                                   float* __restrict__ out) {
    __shared__ float wev[MM * DD];
    __shared__ float red[6];
    int t = threadIdx.x;
    if (t < MM * DD) wev[t] = w_eval[t];
    if (t < 6) red[t] = 0.0f;
    __syncthreads();

    int n = blockIdx.x * 256 + t;
    float acc[6] = {0, 0, 0, 0, 0, 0};
    if (n < NTOT) {
        float pc = pos_cnt[n], nc = neg_cnt[n];
        float cnt = pc + nc;
        if (cnt > 0.0f) {
            float tp = scal[0], tn = scal[1];
            float pw = (tp > 0.0f) ? (tn / fmaxf(tp, 1.0f)) : 1.0f;
            float gold = pc / fmaxf(cnt, 1.0f);
#pragma unroll
            for (int m = 0; m < MM; ++m) {
                const uint4* sp = (const uint4*)(store + ((size_t)n * MM + m) * DD);
                float v = 0.0f;
#pragma unroll
                for (int qv = 0; qv < 16; ++qv) {
                    uint4 u = sp[qv];
                    unsigned int w[4] = {u.x, u.y, u.z, u.w};
#pragma unroll
                    for (int j = 0; j < 4; ++j) {
                        int d = qv * 8 + j * 2;
                        v += bf2f((unsigned short)(w[j] & 0xFFFFu)) * wev[m * DD + d];
                        v += bf2f((unsigned short)(w[j] >> 16))     * wev[m * DD + d + 1];
                    }
                }
                v += b_eval[m];
                float lg  = log1pf(expf(-fabsf(v)));
                float spn = fmaxf(-v, 0.0f) + lg;   // softplus(-v)
                float spp = fmaxf(v, 0.0f) + lg;    // softplus(v)
                float bce = pw * gold * spn + (1.0f - gold) * spp;
                acc[m]     += cnt * bce;
                acc[2 + m] += (v >= 0.0f) ? pc : 0.0f;
                acc[4 + m] += (v < 0.0f) ? nc : 0.0f;
            }
        }
    }
#pragma unroll
    for (int i = 0; i < 6; ++i)
        if (acc[i] != 0.0f) atomicAdd(&red[i], acc[i]);
    __syncthreads();
    if (t < 6) atomicAdd(&out[t], red[t]);
}

extern "C" void kernel_launch(void* const* d_in, const int* in_sizes, int n_in,
                              void* d_out, int out_size, void* d_ws, size_t ws_size,
                              hipStream_t stream) {
    (void)in_sizes; (void)n_in; (void)out_size; (void)ws_size;
    const int*   thax_ids   = (const int*)d_in[0];
    const int*   sine_ids   = (const int*)d_in[1];
    const int*   parents    = (const int*)d_in[2];
    const int*   rule_ids   = (const int*)d_in[3];
    const float* pos_cnt    = (const float*)d_in[4];
    const float* neg_cnt    = (const float*)d_in[5];
    const float* thax_table = (const float*)d_in[6];
    const float* sine_w     = (const float*)d_in[7];
    const float* sine_b     = (const float*)d_in[8];
    const float* W1         = (const float*)d_in[9];
    const float* b1         = (const float*)d_in[10];
    const float* W2         = (const float*)d_in[11];
    const float* b2         = (const float*)d_in[12];
    const float* w_eval     = (const float*)d_in[13];
    const float* b_eval     = (const float*)d_in[14];
    float* out = (float*)d_out;

    // workspace carve-up (256B aligned)
    char* ws = (char*)d_ws;
    size_t off = 0;
    auto carve = [&](size_t bytes) -> char* {
        char* p = ws + off;
        off = (off + bytes + 255) & ~(size_t)255;
        return p;
    };
    unsigned short* store = (unsigned short*)carve((size_t)NTOT * MM * DD * 2);      // 71 MB bf16
    unsigned short* W1t   = (unsigned short*)carve((size_t)MM * RR * 2 * DD * DD * 2);
    unsigned short* W2t   = (unsigned short*)carve((size_t)MM * RR * DD * DD * 2);
    int*            sorted = (int*)carve((size_t)LL * KPAD * 4);
    int*            trule  = (int*)carve((size_t)LL * TMAX * 4);
    float*          scal   = (float*)carve(2 * sizeof(float));

    hipMemsetAsync(out, 0, 6 * sizeof(float), stream);
    hipMemsetAsync(scal, 0, 2 * sizeof(float), stream);

    cvt_w1_kernel<<<(MM * RR * 2 * DD * DD + 255) / 256, 256, 0, stream>>>(W1, W1t);
    cvt_w2_kernel<<<(MM * RR * DD * DD + 255) / 256, 256, 0, stream>>>(W2, W2t);
    init_kernel<<<(N0C * DD + 255) / 256, 256, 0, stream>>>(thax_ids, sine_ids,
                                                            thax_table, sine_w, sine_b, store);
    bucket_kernel<<<LL, 256, 0, stream>>>(rule_ids, sorted, trule);

    dim3 lgrid(TMAX, MM);
    for (int l = 0; l < LL; ++l) {
        level_kernel<<<lgrid, 128, 0, stream>>>(store, W1t, W2t, b1, b2,
                                                sorted, trule, parents, l);
    }

    sums_kernel<<<(NTOT + 255) / 256, 256, 0, stream>>>(pos_cnt, neg_cnt, scal);
    eval_kernel<<<(NTOT + 255) / 256, 256, 0, stream>>>(store, w_eval, b_eval,
                                                        pos_cnt, neg_cnt, scal, out);
}
